// InceptionAttention_11089605558726
// MI455X (gfx1250) — matrix operand order repaired and checked by reading, not compiled, not run on hardware
//
#include <hip/hip_runtime.h>
#include <math.h>

typedef __attribute__((ext_vector_type(16))) _Float16 v16h;
typedef __attribute__((ext_vector_type(8)))  float    v8f;

#define DIMC  256
#define HW    4096     // 64*64
#define NHEAD 8
#define DHEAD 32
#define IW    65       // integral grid edge (65x65, zero row/col at 0)

__device__ __forceinline__ float sigmoidf_(float z) {
  return 1.0f / (1.0f + __expf(-z));
}

// ---------------------------------------------------------------------------
// GEMM1: q/k/v[o,p] = sum_c W[o,c] * x[c,p]; sigmoid fused on q,k.
// One wave = one 16x16 output tile, 3 WMMA accumulators share one B (x) tile.
// A layout (16x32 f16, ISA 7.12.2): lane<16 -> K {0..7,16..23}, lane>=16 -> {8..15,24..31}
// B layout (32x16 f16): same K order as A: lane<16 -> K {0..7,16..23}, lane>=16 -> {8..15,24..31}
// C/D (16x16 f32): vgpr j, lane l -> M = j + (l>>4)*8, N = l&15
// ---------------------------------------------------------------------------
__global__ __launch_bounds__(256)
void gemm_qkv_kernel(const float* __restrict__ x,
                     const float* __restrict__ Wq,
                     const float* __restrict__ Wk,
                     const float* __restrict__ Wv,
                     float* __restrict__ q,
                     float* __restrict__ k,
                     float* __restrict__ v)
{
  const int lane = threadIdx.x & 31;
  const int wave = threadIdx.x >> 5;
  const int tile = blockIdx.x * 8 + wave;        // 16 Mtiles * 256 Ntiles = 4096
  const int m0   = (tile >> 8) * 16;
  const int n0   = (tile & 255) * 16;
  const int half = lane >> 4;
  const int l15  = lane & 15;
  const int arow = m0 + l15;
  const int p    = n0 + l15;

  v8f accq = {}, acck = {}, accv = {};
  for (int kb = 0; kb < DIMC; kb += 32) {
    v16h Aq, Ak, Av, Bx;
#pragma unroll
    for (int i = 0; i < 16; ++i) {
      const int kk = kb + half * 8 + ((i & 8) << 1) + (i & 7);
      Aq[i] = (_Float16)Wq[arow * DIMC + kk];
      Ak[i] = (_Float16)Wk[arow * DIMC + kk];
      Av[i] = (_Float16)Wv[arow * DIMC + kk];
    }
#pragma unroll
    for (int i = 0; i < 16; ++i) {
      const int kk = kb + half * 8 + ((i & 8) << 1) + (i & 7);
      Bx[i] = (_Float16)x[kk * HW + p];
    }
    if (kb + 32 < DIMC) __builtin_prefetch(&x[(kb + 32) * HW + p], 0, 1);
    accq = __builtin_amdgcn_wmma_f32_16x16x32_f16(false, Aq, false, Bx, (short)0, accq, false, false);
    acck = __builtin_amdgcn_wmma_f32_16x16x32_f16(false, Ak, false, Bx, (short)0, acck, false, false);
    accv = __builtin_amdgcn_wmma_f32_16x16x32_f16(false, Av, false, Bx, (short)0, accv, false, false);
  }
#pragma unroll
  for (int j = 0; j < 8; ++j) {
    const int m = m0 + j + half * 8;
    q[m * HW + p] = sigmoidf_(accq[j]);
    k[m * HW + p] = sigmoidf_(acck[j]);
    v[m * HW + p] = accv[j];
  }
}

// ---------------------------------------------------------------------------
// v <- v / max(||v||_head, 1e-12), per (head, pixel). Coalesced across pixels.
// ---------------------------------------------------------------------------
__global__ __launch_bounds__(256)
void vnorm_kernel(float* __restrict__ v)
{
  const int t = blockIdx.x * blockDim.x + threadIdx.x;   // 8*4096
  if (t >= NHEAD * HW) return;
  const int p = t & (HW - 1);
  const int c0 = (t >> 12) * DHEAD;
  float ss = 0.f;
#pragma unroll
  for (int d = 0; d < DHEAD; ++d) {
    const float a = v[(c0 + d) * HW + p];
    ss += a * a;
  }
  const float inv = 1.0f / fmaxf(sqrtf(ss), 1e-12f);
#pragma unroll
  for (int d = 0; d < DHEAD; ++d) v[(c0 + d) * HW + p] *= inv;
}

// ---------------------------------------------------------------------------
// Ik: [65][65][256], zero row/col 0, Ik[y+1][x+1][c] = k[c][p]; then 2 cumsums.
// ---------------------------------------------------------------------------
__global__ __launch_bounds__(256)
void ik_fill_kernel(const float* __restrict__ k, float* __restrict__ Ik)
{
  const int total = IW * IW * DIMC;
  for (int idx = blockIdx.x * blockDim.x + threadIdx.x; idx < total;
       idx += gridDim.x * blockDim.x) {
    const int c  = idx & 255;
    const int xy = idx >> 8;
    const int X  = xy % IW, Y = xy / IW;
    float val = 0.f;
    if (X > 0 && Y > 0) val = k[c * HW + (Y - 1) * 64 + (X - 1)];
    Ik[idx] = val;
  }
}

__global__ __launch_bounds__(256)
void ik_cumsum_x_kernel(float* __restrict__ Ik)
{
  const int t = blockIdx.x * blockDim.x + threadIdx.x;   // 64*256
  if (t >= 64 * DIMC) return;
  const int c = t & 255;
  const int y = 1 + (t >> 8);
  float run = Ik[(y * IW + 1) * DIMC + c];
  for (int x = 2; x < IW; ++x) {
    const int i = (y * IW + x) * DIMC + c;
    run += Ik[i];
    Ik[i] = run;
  }
}

__global__ __launch_bounds__(256)
void ik_cumsum_y_kernel(float* __restrict__ Ik)
{
  const int t = blockIdx.x * blockDim.x + threadIdx.x;   // 64*256
  if (t >= 64 * DIMC) return;
  const int c = t & 255;
  const int x = 1 + (t >> 8);
  float run = Ik[(1 * IW + x) * DIMC + c];
  for (int y = 2; y < IW; ++y) {
    const int i = (y * IW + x) * DIMC + c;
    run += Ik[i];
    Ik[i] = run;
  }
}

// ---------------------------------------------------------------------------
// Per-head Ikv: [65][65][32(dk)][32(gv)]; 17.3 MB -> stays resident in the
// 192 MB L2 so the attn corner gathers never touch HBM.
// ---------------------------------------------------------------------------
__global__ __launch_bounds__(256)
void ikv_fill_kernel(const float* __restrict__ k, const float* __restrict__ v,
                     float* __restrict__ Ikv, int head)
{
  const int total = IW * IW * DHEAD * DHEAD;
  const int c0 = head * DHEAD;
  for (int idx = blockIdx.x * blockDim.x + threadIdx.x; idx < total;
       idx += gridDim.x * blockDim.x) {
    const int dg = idx & 1023;
    const int xy = idx >> 10;
    const int X  = xy % IW, Y = xy / IW;
    float val = 0.f;
    if (X > 0 && Y > 0) {
      const int p  = (Y - 1) * 64 + (X - 1);
      const int dk = dg >> 5, gv = dg & 31;
      val = k[(c0 + dk) * HW + p] * v[(c0 + gv) * HW + p];
    }
    Ikv[idx] = val;
  }
}

__global__ __launch_bounds__(256)
void ikv_cumsum_x_kernel(float* __restrict__ Ikv)
{
  const int t = blockIdx.x * blockDim.x + threadIdx.x;   // 64*1024
  if (t >= 64 * 1024) return;
  const int dg = t & 1023;
  const int y  = 1 + (t >> 10);
  float run = Ikv[(y * IW + 1) * 1024 + dg];
  for (int x = 2; x < IW; ++x) {
    const int i = (y * IW + x) * 1024 + dg;
    run += Ikv[i];
    Ikv[i] = run;
  }
}

__global__ __launch_bounds__(256)
void ikv_cumsum_y_kernel(float* __restrict__ Ikv)
{
  const int t = blockIdx.x * blockDim.x + threadIdx.x;   // 64*1024
  if (t >= 64 * 1024) return;
  const int dg = t & 1023;
  const int x  = 1 + (t >> 10);
  float run = Ikv[(1 * IW + x) * 1024 + dg];
  for (int y = 2; y < IW; ++y) {
    const int i = (y * IW + x) * 1024 + dg;
    run += Ikv[i];
    Ikv[i] = run;
  }
}

// ---------------------------------------------------------------------------
// Attention: one wave32 per pixel, lane = output channel f (and d for norm).
// Corners: y1 = clamp(y-ws/2), y2 = clamp(y+ws/2) (== the reference's
// replicate-padded integral box diff for ws in {64,32,16}).
// blockIdx.y = window index.
// ---------------------------------------------------------------------------
__global__ __launch_bounds__(256)
void attn_kernel(const float* __restrict__ q,
                 const float* __restrict__ Ik,    // [65][65][256]
                 const float* __restrict__ Ikv,   // this head: [65][65][32][32]
                 float* __restrict__ o,           // [768][4096]
                 int head)
{
  const int lane = threadIdx.x & 31;
  const int wave = threadIdx.x >> 5;
  const int p    = blockIdx.x * 8 + wave;          // 512 blocks * 8 waves = 4096
  const int widx = blockIdx.y;                     // 0..2 -> ws 64,32,16
  const int hw   = (64 >> widx) >> 1;
  const int y = p >> 6, x = p & 63;
  const int y1 = max(y - hw, 0), y2 = min(y + hw, 64);
  const int x1 = max(x - hw, 0), x2 = min(x + hw, 64);
  const int c0 = head * DHEAD;

  const float qown = q[(c0 + lane) * HW + p];

  // norm = sum_d q_d * box(Ik)_d   (lane d, wave reduction)
  const int b11 = (y1 * IW + x1) * DIMC + c0 + lane;
  const int b12 = (y1 * IW + x2) * DIMC + c0 + lane;
  const int b21 = (y2 * IW + x1) * DIMC + c0 + lane;
  const int b22 = (y2 * IW + x2) * DIMC + c0 + lane;
  float part = qown * (Ik[b22] - Ik[b21] - Ik[b12] + Ik[b11]);
#pragma unroll
  for (int off = 16; off; off >>= 1) part += __shfl_xor(part, off, 32);
  const float invn = 1.0f / (part + 1e-6f);

  // attn_f = sum_d q_d * box(Ikv)_{d,f}   (lane f, coalesced corner rows)
  const float* r11 = Ikv + (y1 * IW + x1) * 1024;
  const float* r12 = Ikv + (y1 * IW + x2) * 1024;
  const float* r21 = Ikv + (y2 * IW + x1) * 1024;
  const float* r22 = Ikv + (y2 * IW + x2) * 1024;
  float acc = 0.f;
#pragma unroll 4
  for (int d = 0; d < DHEAD; ++d) {
    const float qd = __shfl(qown, d, 32);
    const int   off = d * 32 + lane;
    acc += qd * (r22[off] - r21[off] - r12[off] + r11[off]);
  }
  o[(widx * DIMC + c0 + lane) * HW + p] = acc * invn;
}

// ---------------------------------------------------------------------------
// GEMM2: out[m,p] = sum_j Wout[m,j] * o[j,p] + bout[m], K = 768, WMMA f16.
// ---------------------------------------------------------------------------
__global__ __launch_bounds__(256)
void gemm_out_kernel(const float* __restrict__ Wout,   // (256, 768)
                     const float* __restrict__ o,      // (768, 4096)
                     const float* __restrict__ bout,
                     float* __restrict__ out)
{
  const int lane = threadIdx.x & 31;
  const int wave = threadIdx.x >> 5;
  const int tile = blockIdx.x * 8 + wave;
  const int m0   = (tile >> 8) * 16;
  const int n0   = (tile & 255) * 16;
  const int half = lane >> 4;
  const int l15  = lane & 15;
  const int arow = m0 + l15;
  const int p    = n0 + l15;
  const int K    = 3 * DIMC;   // 768

  v8f acc = {};
  for (int kb = 0; kb < K; kb += 32) {
    v16h A, B;
#pragma unroll
    for (int i = 0; i < 16; ++i) {
      const int kk = kb + half * 8 + ((i & 8) << 1) + (i & 7);
      A[i] = (_Float16)Wout[arow * K + kk];
    }
#pragma unroll
    for (int i = 0; i < 16; ++i) {
      const int kk = kb + half * 8 + ((i & 8) << 1) + (i & 7);
      B[i] = (_Float16)o[kk * HW + p];
    }
    if (kb + 32 < K) __builtin_prefetch(&o[(kb + 32) * HW + p], 0, 1);
    acc = __builtin_amdgcn_wmma_f32_16x16x32_f16(false, A, false, B, (short)0, acc, false, false);
  }
#pragma unroll
  for (int j = 0; j < 8; ++j) {
    const int m = m0 + j + half * 8;
    out[m * HW + p] = acc[j] + bout[m];
  }
}

// ---------------------------------------------------------------------------
extern "C" void kernel_launch(void* const* d_in, const int* in_sizes, int n_in,
                              void* d_out, int out_size, void* d_ws, size_t ws_size,
                              hipStream_t stream)
{
  const float* x    = (const float*)d_in[0];
  const float* Wq   = (const float*)d_in[1];
  const float* Wk   = (const float*)d_in[2];
  const float* Wv   = (const float*)d_in[3];
  const float* Wout = (const float*)d_in[4];
  const float* bout = (const float*)d_in[5];
  float* out = (float*)d_out;

  // Workspace layout (floats): q,k,v (1M each) | Ik 1.08M | Ikv 4.33M | o 3.15M
  float* q   = (float*)d_ws;
  float* k   = q  + DIMC * HW;
  float* v   = k  + DIMC * HW;
  float* Ik  = v  + DIMC * HW;
  float* Ikv = Ik + IW * IW * DIMC;
  float* o   = Ikv + IW * IW * DHEAD * DHEAD;

  gemm_qkv_kernel<<<512, 256, 0, stream>>>(x, Wq, Wk, Wv, q, k, v);
  vnorm_kernel<<<(NHEAD * HW + 255) / 256, 256, 0, stream>>>(v);

  ik_fill_kernel<<<1024, 256, 0, stream>>>(k, Ik);
  ik_cumsum_x_kernel<<<64, 256, 0, stream>>>(Ik);
  ik_cumsum_y_kernel<<<64, 256, 0, stream>>>(Ik);

  for (int h = 0; h < NHEAD; ++h) {
    ikv_fill_kernel<<<4096, 256, 0, stream>>>(k, v, Ikv, h);
    ikv_cumsum_x_kernel<<<256, 256, 0, stream>>>(Ikv);
    ikv_cumsum_y_kernel<<<256, 256, 0, stream>>>(Ikv);
    attn_kernel<<<dim3(512, 3), 256, 0, stream>>>(q, Ik, Ikv, o, h);
  }

  gemm_out_kernel<<<512, 256, 0, stream>>>(Wout, o, bout, out);
}
